// BatchMuSc_65678639891090
// MI455X (gfx1250) — matrix operand order repaired and checked by reading, not compiled, not run on hardware
//
#include <hip/hip_runtime.h>

typedef __attribute__((ext_vector_type(16))) __bf16 v16bf;
typedef __attribute__((ext_vector_type(8)))  float  v8f;
typedef __attribute__((ext_vector_type(4)))  int    v4i;

#define NIMG   16
#define LROW   1369
#define CDIM   1024
#define NLTOT  (NIMG * LROW)          // 21904 patches total
#define RT     128                    // A rows per workgroup
#define NTILES 11                     // ceil(1369/128) row tiles
#define NCB    11                     // ceil(1369/128) column blocks
#define KC     64                     // B K-chunk staged in LDS
#define NKC    (CDIM / KC)            // 16 K-chunks
#define ASTR   (CDIM + 8)             // 1032 ushorts: 2064B stride == 1 mod 16 slots
#define BSTR   (KC + 8)               // 72 ushorts:   144B stride == 9 mod 16 slots
#define A_USH  ((size_t)RT * ASTR)    // 132,096 ushorts (264,192 B) persistent A tile
#define B_USH  ((size_t)RT * BSTR)    //   9,216 ushorts ( 18,432 B) per B chunk buffer

#if __has_builtin(__builtin_amdgcn_global_load_async_to_lds_b128) && \
    __has_builtin(__builtin_amdgcn_s_wait_asynccnt)
#define HAVE_ASYNC 1
#else
#define HAVE_ASYNC 0
#endif

struct FragBits { int4 lo, hi; };     // 32 bytes == sizeof(v16bf)

static __device__ __forceinline__ v16bf make_frag(const unsigned short* p0,
                                                  const unsigned short* p1) {
  FragBits t;
  t.lo = *(const int4*)p0;
  t.hi = *(const int4*)p1;
  return __builtin_bit_cast(v16bf, t);
}

// ---------------------------------------------------------------------------
// Kernel 1: f32 -> bf16 (RNE) conversion + per-patch squared norm (f32).
// ---------------------------------------------------------------------------
__global__ void __launch_bounds__(256) k_prep(const float* __restrict__ Z,
                                              unsigned short* __restrict__ Zbf,
                                              float* __restrict__ norms) {
  const int p = blockIdx.x;                       // patch id in [0, NLTOT)
  const float* src = Z + (size_t)p * CDIM;
  unsigned short* dst = Zbf + (size_t)p * CDIM;
  float ss = 0.0f;
  for (int e = threadIdx.x; e < CDIM; e += 256) {
    float x = src[e];
    ss += x * x;
    unsigned u = __float_as_uint(x);
    unsigned r = (u + 0x7FFFu + ((u >> 16) & 1u)) >> 16;   // round-to-nearest-even
    dst[e] = (unsigned short)r;
  }
  const int lane = threadIdx.x & 31, wv = threadIdx.x >> 5;
  #pragma unroll
  for (int m = 16; m >= 1; m >>= 1) ss += __shfl_xor(ss, m, 32);
  __shared__ float red[8];
  if (lane == 0) red[wv] = ss;
  __syncthreads();
  if (threadIdx.x == 0) {
    float t = red[0];
    #pragma unroll
    for (int k = 1; k < 8; ++k) t += red[k];
    norms[p] = t;
  }
}

// ---------------------------------------------------------------------------
// Kernel 2: fused bf16-WMMA GEMM + column-min.
//   minmat[i, l, j] = min_m sqrt(max(|a_l|^2 + |b_m|^2 - 2 a_l.b_m, 1e-12))
// Grid (row-tile, j, i); 256 threads = 8 wave32 arranged 4 Mw x 2 Nw.
// A tile (128 x 1024 bf16, 264 KB) is LDS-persistent; B is staged in
// 128-col x 64-K chunks via GLOBAL_LOAD_ASYNC_TO_LDS_B128 with THREE
// chunk buffers (2-deep ASYNCcnt pipeline); the kc loop is fully unrolled
// so accumulators stay in fixed registers (no phi copies / hazard NOPs).
// Each wave owns a 32x64 output tile (MB=2 x NB=4): 12 ds_read_b128 per
// 8 WMMAs.
// ---------------------------------------------------------------------------
__global__ void __launch_bounds__(256, 1) k_distmin(const unsigned short* __restrict__ Zbf,
                                                    const float* __restrict__ norms,
                                                    float* __restrict__ minmat) {
  extern __shared__ unsigned short smem[];
  unsigned short* As = smem;                      // persistent A tile: RT x ASTR
  unsigned short* Bs = smem + A_USH;              // B chunks: 3 x (RT x BSTR)
  float* red = (float*)smem;                      // aliased reduction buffer [128][2]

  const int rt = blockIdx.x, j = blockIdx.y, i = blockIdx.z;
  const int rowBase = rt * RT;

  if (i == j) {                                   // self image: distance = +inf
    for (int r = threadIdx.x; r < RT; r += 256) {
      int l = rowBase + r;
      if (l < LROW) minmat[((size_t)i * LROW + l) * NIMG + j] = __builtin_inff();
    }
    return;
  }

  const int t    = threadIdx.x;
  const int lane = t & 31, wv = t >> 5;
  const int mw   = wv >> 1;                       // 0..3: which 32-row strip
  const int nw   = wv & 1;                        // 0..1: which 64-col strip
  const int nlo  = lane & 15;
  const bool hi  = lane >= 16;
  const int tr8  = t >> 3, tc8 = t & 7;           // B staging: row / 16B chunk (8/row)

  const size_t aBaseG = (size_t)i * LROW * CDIM;
  const size_t bBaseG = (size_t)j * LROW * CDIM;

  // Stage the entire A tile once (128 rows x 1024 K bf16).
  for (int idx = t; idx < RT * (CDIM / 8); idx += 256) {
    int row = idx >> 7, ch = idx & 127;           // 128 int4 chunks per row
    int l = rowBase + row; if (l >= LROW) l = LROW - 1;
    *(int4*)(As + (size_t)row * ASTR + ch * 8) =
        *(const int4*)(Zbf + aBaseG + (size_t)l * CDIM + ch * 8);
  }

  float rmin[2][8];
  #pragma unroll
  for (int mt = 0; mt < 2; ++mt)
    #pragma unroll
    for (int r = 0; r < 8; ++r) rmin[mt][r] = __builtin_inff();

  for (int cb = 0; cb < NCB; ++cb) {
    v8f acc[2][4] = {};

    // One K-chunk of WMMAs against B chunk buffer `Bbuf`.
    auto mma_chunk = [&](const unsigned short* Bbuf, int kc) {
      // ISA 7.12.2 bf16 layouts:
      //  A (16x32): lane<16 -> K {0..7,16..23}; lane>=16 -> K {8..15,24..31}, M = lane%16
      //  B (32x16): lane<16 -> K {0..7,16..23}; lane>=16 -> K {8..15,24..31}, N = lane%16
      const unsigned short* aB = As + (size_t)(mw * 32 + nlo) * ASTR + kc * KC + (hi ? 8 : 0);
      const unsigned short* bB = Bbuf + (size_t)(nw * 64 + nlo) * BSTR + (hi ? 8 : 0);
      #pragma unroll
      for (int kt = 0; kt < KC / 32; ++kt) {
        v16bf fa[2], fb[4];
        #pragma unroll
        for (int mt = 0; mt < 2; ++mt) {
          const unsigned short* p = aB + (size_t)mt * 16 * ASTR + kt * 32;
          fa[mt] = make_frag(p, p + 16);          // two 16B runs (K, K+16)
        }
        #pragma unroll
        for (int nt = 0; nt < 4; ++nt) {
          const unsigned short* p = bB + (size_t)nt * 16 * BSTR + kt * 32;
          fb[nt] = make_frag(p, p + 16);          // two 16B runs (K, K+16)
        }
        #pragma unroll
        for (int mt = 0; mt < 2; ++mt)
          #pragma unroll
          for (int nt = 0; nt < 4; ++nt)
            acc[mt][nt] = __builtin_amdgcn_wmma_f32_16x16x32_bf16(
                false, fa[mt], false, fb[nt], (short)0, acc[mt][nt], false, false);
      }
    };

#if HAVE_ASYNC
    // Async global->LDS (no staging VGPRs), triple-buffered chunks,
    // two chunks in flight.
    auto issue = [&](int kc, int buf) {
      unsigned short* Bb = Bs + (size_t)buf * B_USH;
      #pragma unroll
      for (int q = 0; q < 4; ++q) {
        int m = cb * 128 + q * 32 + tr8; if (m >= LROW) m = LROW - 1;
        const unsigned short* src = Zbf + bBaseG + (size_t)m * CDIM + kc * KC + tc8 * 8;
        unsigned short* dst = Bb + (size_t)(q * 32 + tr8) * BSTR + tc8 * 8;
        __builtin_amdgcn_global_load_async_to_lds_b128(
            (__attribute__((address_space(1))) v4i*)src,
            (__attribute__((address_space(3))) v4i*)dst, 0, 0);
      }
    };
    issue(0, 0);
    issue(1, 1);
    #pragma unroll
    for (int kc = 0; kc < NKC; ++kc) {
      if (kc + 2 < NKC) {
        issue(kc + 2, (kc + 2) % 3);              // prefetch two chunks ahead
        __builtin_amdgcn_s_wait_asynccnt(8);      // drain chunk kc, keep kc+1/kc+2
      } else if (kc + 1 < NKC) {
        __builtin_amdgcn_s_wait_asynccnt(4);      // drain chunk kc, keep kc+1
      } else {
        __builtin_amdgcn_s_wait_asynccnt(0);
      }
      __syncthreads();                            // all waves' chunk-kc loads landed
      mma_chunk(Bs + (size_t)(kc % 3) * B_USH, kc);
      __syncthreads();                            // buffer free before chunk kc+3 writes
    }
#else
    // Sync fallback: small 4-int4 register staging (no spills).
    int4 stg[4];
    auto gload = [&](int kc) {
      #pragma unroll
      for (int q = 0; q < 4; ++q) {
        int m = cb * 128 + q * 32 + tr8; if (m >= LROW) m = LROW - 1;
        stg[q] = *(const int4*)(Zbf + bBaseG + (size_t)m * CDIM + kc * KC + tc8 * 8);
      }
    };
    auto commit = [&]() {
      #pragma unroll
      for (int q = 0; q < 4; ++q)
        *(int4*)(Bs + (size_t)(q * 32 + tr8) * BSTR + tc8 * 8) = stg[q];
    };
    gload(0);
    for (int kc = 0; kc < NKC; ++kc) {
      __syncthreads();                            // B buffer free (prev compute done)
      commit();
      __syncthreads();                            // chunk (and, first time, A) visible
      if (kc + 1 < NKC) gload(kc + 1);            // prefetch next chunk into regs
      mma_chunk(Bs, kc);
    }
#endif

    // Fold |b_m|^2 - 2*dot into the running per-row min for this column block.
    #pragma unroll
    for (int nt = 0; nt < 4; ++nt) {
      int m = cb * 128 + nw * 64 + nt * 16 + nlo;
      float nb = (m < LROW) ? norms[(size_t)j * LROW + m] : __builtin_inff();
      #pragma unroll
      for (int mt = 0; mt < 2; ++mt)
        #pragma unroll
        for (int r = 0; r < 8; ++r)
          rmin[mt][r] = fminf(rmin[mt][r], __builtin_fmaf(-2.0f, acc[mt][nt][r], nb));
    }
  }

  // Min across the 16 lanes of each half (lanes 0-15: rows r, 16-31: rows r+8).
  #pragma unroll
  for (int mt = 0; mt < 2; ++mt)
    #pragma unroll
    for (int r = 0; r < 8; ++r) {
      float v = rmin[mt][r];
      #pragma unroll
      for (int m = 8; m >= 1; m >>= 1) v = fminf(v, __shfl_xor(v, m, 32));
      rmin[mt][r] = v;
    }

  __syncthreads();                                // done with LDS tiles; alias as red[]
  if (nlo == 0) {
    #pragma unroll
    for (int mt = 0; mt < 2; ++mt)
      #pragma unroll
      for (int r = 0; r < 8; ++r) {
        int row = mw * 32 + mt * 16 + (hi ? 8 : 0) + r;   // 0..127 within WG tile
        red[row * 2 + nw] = rmin[mt][r];
      }
  }
  __syncthreads();
  if (t < RT) {
    int l = rowBase + t;
    if (l < LROW) {
      float v = fminf(red[t * 2 + 0], red[t * 2 + 1]);
      float na = norms[(size_t)i * LROW + l];
      minmat[((size_t)i * LROW + l) * NIMG + j] = __builtin_sqrtf(fmaxf(na + v, 1e-12f));
    }
  }
}

// ---------------------------------------------------------------------------
// Kernel 3: per patch, mean of the 4 smallest per-image min-distances
// (k_max = int(15*0.3) = 4, k_min = 0; self column is +inf).
// ---------------------------------------------------------------------------
__global__ void __launch_bounds__(256) k_score(const float* __restrict__ minmat,
                                               float* __restrict__ pscores) {
  int t = blockIdx.x * 256 + threadIdx.x;
  if (t >= NLTOT) return;
  const float* row = minmat + (size_t)t * NIMG;
  float s0 = __builtin_inff(), s1 = s0, s2 = s0, s3 = s0;
  #pragma unroll
  for (int j = 0; j < NIMG; ++j) {
    float v = row[j];
    if (v < s3) {
      if      (v < s0) { s3 = s2; s2 = s1; s1 = s0; s0 = v; }
      else if (v < s1) { s3 = s2; s2 = s1; s1 = v; }
      else if (v < s2) { s3 = s2; s2 = v; }
      else             { s3 = v; }
    }
  }
  pscores[t] = 0.25f * (s0 + s1 + s2 + s3);
}

// ---------------------------------------------------------------------------
// Kernel 4: per-image max over 1369 patch scores.
// ---------------------------------------------------------------------------
__global__ void __launch_bounds__(256) k_imgmax(const float* __restrict__ pscores,
                                                float* __restrict__ img) {
  const int i = blockIdx.x;
  float mx = -__builtin_inff();
  for (int l = threadIdx.x; l < LROW; l += 256)
    mx = fmaxf(mx, pscores[(size_t)i * LROW + l]);
  const int lane = threadIdx.x & 31, wv = threadIdx.x >> 5;
  #pragma unroll
  for (int m = 16; m >= 1; m >>= 1) mx = fmaxf(mx, __shfl_xor(mx, m, 32));
  __shared__ float red[8];
  if (lane == 0) red[wv] = mx;
  __syncthreads();
  if (threadIdx.x == 0) {
    float t = red[0];
    #pragma unroll
    for (int k = 1; k < 8; ++k) t = fmaxf(t, red[k]);
    img[i] = t;
  }
}

// ---------------------------------------------------------------------------
// Kernel 5: min-max normalize, W = cls @ cls^T, MMO over k = {1,2,3}.
// ---------------------------------------------------------------------------
__global__ void __launch_bounds__(256) k_final(const float* __restrict__ cls,
                                               const float* __restrict__ img,
                                               float* __restrict__ out) {
  __shared__ float Wm[NIMG][NIMG];
  __shared__ float sv[NIMG];
  const int t = threadIdx.x;
  const int r = t >> 4, c = t & 15;
  {
    const float* ar = cls + (size_t)r * CDIM;
    const float* bc = cls + (size_t)c * CDIM;
    float d = 0.0f;
    for (int k = 0; k < CDIM; ++k) d = __builtin_fmaf(ar[k], bc[k], d);
    Wm[r][c] = d;
  }
  if (t < NIMG) {
    float mn = __builtin_inff(), mx = -__builtin_inff();
    #pragma unroll
    for (int k = 0; k < NIMG; ++k) { mn = fminf(mn, img[k]); mx = fmaxf(mx, img[k]); }
    sv[t] = (img[t] - mn) / (mx - mn);
  }
  __syncthreads();
  if (t < NIMG) {
    float w[NIMG];
    #pragma unroll
    for (int c2 = 0; c2 < NIMG; ++c2) w[c2] = Wm[t][c2];
    float t1 = -__builtin_inff(), t2 = t1, t3 = t1;          // top-3 values
    #pragma unroll
    for (int c2 = 0; c2 < NIMG; ++c2) {
      float v = w[c2];
      if      (v > t1) { t3 = t2; t2 = t1; t1 = v; }
      else if (v > t2) { t3 = t2; t2 = v; }
      else if (v > t3) { t3 = v; }
    }
    float thr[3] = { t1, t2, t3 };
    float accv = 0.0f;
    #pragma unroll
    for (int k = 0; k < 3; ++k) {
      float ssum = 0.0f, sdot = 0.0f;
      #pragma unroll
      for (int c2 = 0; c2 < NIMG; ++c2) {
        if (w[c2] >= thr[k]) { ssum += w[c2]; sdot += w[c2] * sv[c2]; }
      }
      accv += sdot / ssum;
    }
    out[t] = accv * (1.0f / 3.0f);
  }
}

// ---------------------------------------------------------------------------
extern "C" void kernel_launch(void* const* d_in, const int* in_sizes, int n_in,
                              void* d_out, int out_size, void* d_ws, size_t ws_size,
                              hipStream_t stream) {
  (void)in_sizes; (void)n_in; (void)out_size; (void)ws_size;
  const float* Z   = (const float*)d_in[0];   // [16, 1369, 1024] f32
  const float* cls = (const float*)d_in[1];   // [16, 1024] f32
  float* out = (float*)d_out;                 // [16] f32

  char* ws = (char*)d_ws;
  size_t off = 0;
  auto carve = [&](size_t bytes) -> void* {
    void* p = ws + off;
    off = (off + bytes + 255) & ~(size_t)255;
    return p;
  };
  unsigned short* Zbf  = (unsigned short*)carve((size_t)NLTOT * CDIM * sizeof(unsigned short)); // 44.9 MB
  float* norms   = (float*)carve((size_t)NLTOT * sizeof(float));
  float* minmat  = (float*)carve((size_t)NLTOT * NIMG * sizeof(float));
  float* pscores = (float*)carve((size_t)NLTOT * sizeof(float));
  float* img     = (float*)carve(NIMG * sizeof(float));

  k_prep<<<NLTOT, 256, 0, stream>>>(Z, Zbf, norms);

  const size_t ldsBytes = (A_USH + 3 * B_USH) * sizeof(unsigned short); // 319,488 B <= 320 KB
  k_distmin<<<dim3(NTILES, NIMG, NIMG), 256, ldsBytes, stream>>>(Zbf, norms, minmat);

  k_score<<<(NLTOT + 255) / 256, 256, 0, stream>>>(minmat, pscores);
  k_imgmax<<<NIMG, 256, 0, stream>>>(pscores, img);
  k_final<<<1, 256, 0, stream>>>(cls, img, out);
}
